// JointAttention_67568425500922
// MI455X (gfx1250) — matrix operand order repaired and checked by reading, not compiled, not run on hardware
//
#include <hip/hip_runtime.h>

#define HEADS    24
#define HEAD_DIM 64
#define HIDDEN   1536
#define BATCH    4

typedef __attribute__((ext_vector_type(16))) _Float16 v16h;
typedef __attribute__((ext_vector_type(8)))  float    v8f;

// ---------------- wave32 lane exchange helpers (ds_bpermute) ----------------
static __device__ __forceinline__ float bperm(float v, int srcLane) {
    return __int_as_float(__builtin_amdgcn_ds_bpermute(srcLane << 2, __float_as_int(v)));
}

// ---------------- WMMA wrapper: D = A(16x32 f16) * B(32x16 f16) + C ---------
static __device__ __forceinline__ v8f wmma_f16(v16h a, v16h b, v8f c) {
    return __builtin_amdgcn_wmma_f32_16x16x32_f16(false, a, false, b, (short)0, c, false, false);
}

// Load f16 A-matrix operand from an LDS row (row-major, 64 halves per key row).
// A-layout: half e -> d = e + 16*hi   (+32*dblk)
static __device__ __forceinline__ v16h loadA(const _Float16* __restrict__ row, int hi, int dblk) {
    const _Float16* p0 = row + dblk * 32 + (hi << 4);
    v16h r;
#pragma unroll
    for (int j = 0; j < 16; ++j) { r[j] = p0[j]; }
    return r;
}

// =====================================================================
// Kernel 1: fused head-split + RMSNorm + 2D RoPE + f16 downcast.
// One wave per (b, n, h) row; lane l owns dims (2l, 2l+1) — exactly one
// RoPE rotation pair (lanes 0-15: row-angle half, lanes 16-31: col-angle half).
// Q is pre-scaled by 1/sqrt(D).
// =====================================================================
__global__ __launch_bounds__(256) void prep_kernel(
    const float* __restrict__ q, const float* __restrict__ k, const float* __restrict__ v,
    const float* __restrict__ qw, const float* __restrict__ kw,
    const int* __restrict__ pStart, const int* __restrict__ pLen, const int* __restrict__ pW,
    _Float16* __restrict__ qws, _Float16* __restrict__ kws, _Float16* __restrict__ vws,
    int N)
{
    const int lane = threadIdx.x & 31;
    const int wrow = blockIdx.x * (blockDim.x >> 5) + (threadIdx.x >> 5);
    const int total = BATCH * N * HEADS;
    if (wrow >= total) return;

    const int b   = wrow / (N * HEADS);
    const int rem = wrow - b * (N * HEADS);
    const int n   = rem / HEADS;
    const int h   = rem - n * HEADS;

    const size_t inBase  = ((size_t)b * N + n) * HIDDEN + h * HEAD_DIM + 2 * lane;
    const size_t outBase = (((size_t)b * HEADS + h) * N + n) * HEAD_DIM + 2 * lane;

    const int start = *pStart, len = *pLen, W = *pW;
    const bool doRope = (len > 0) && (n >= start) && (n < start + len);
    float c = 1.0f, s = 0.0f;
    if (doRope) {
        const int p = n - start;
        const float angBase = (lane < 16) ? (float)(p / W) : (float)(p % W);
        const int j = lane & 15;                       // pair index within half
        const float invf = __powf(10000.0f, -(float)j * (1.0f / 16.0f));
        const float ang = angBase * invf;
        c = __cosf(ang); s = __sinf(ang);
    }

    const float w0q = qw[2 * lane], w1q = qw[2 * lane + 1];
    const float w0k = kw[2 * lane], w1k = kw[2 * lane + 1];

    // ---- Q: rmsnorm -> rope -> *1/sqrt(D) -> f16 ----
    {
        float x0 = q[inBase], x1 = q[inBase + 1];
        float ss = x0 * x0 + x1 * x1;
#pragma unroll
        for (int m = 16; m >= 1; m >>= 1) ss += bperm(ss, lane ^ m);
        const float r = rsqrtf(ss * (1.0f / HEAD_DIM) + 1e-6f);
        const float y0 = x0 * r * w0q, y1 = x1 * r * w1q;
        const float o0 = y0 * c - y1 * s, o1 = y0 * s + y1 * c;
        qws[outBase]     = (_Float16)(o0 * 0.125f);
        qws[outBase + 1] = (_Float16)(o1 * 0.125f);
    }
    // ---- K: rmsnorm -> rope -> f16 ----
    {
        float x0 = k[inBase], x1 = k[inBase + 1];
        float ss = x0 * x0 + x1 * x1;
#pragma unroll
        for (int m = 16; m >= 1; m >>= 1) ss += bperm(ss, lane ^ m);
        const float r = rsqrtf(ss * (1.0f / HEAD_DIM) + 1e-6f);
        const float y0 = x0 * r * w0k, y1 = x1 * r * w1k;
        kws[outBase]     = (_Float16)(y0 * c - y1 * s);
        kws[outBase + 1] = (_Float16)(y0 * s + y1 * c);
    }
    // ---- V: plain downcast ----
    vws[outBase]     = (_Float16)v[inBase];
    vws[outBase + 1] = (_Float16)v[inBase + 1];
}

// =====================================================================
// Kernel 2: flash attention. Block = 128 threads (4 waves), each wave owns a
// 16-query tile; block owns 64 queries of one (b,h). Loop over 32-key chunks:
//   S^T = K_chunk x Q^T  (so P lands directly in the A-operand layout for PV)
//   online softmax with one lane0<->16 exchange + 8 bpermute broadcasts
//   O += P x V           (V staged transposed in LDS for contiguous B reads)
// =====================================================================
__global__ __launch_bounds__(128) void attn_kernel(
    const _Float16* __restrict__ qws, const _Float16* __restrict__ kws,
    const _Float16* __restrict__ vws, const unsigned char* __restrict__ mask,
    float* __restrict__ out, int N)
{
    __shared__ _Float16 Klds[32 * 64];   // [key][d]   row-major, 4KB
    __shared__ _Float16 Vt[64 * 32];     // [d][key]   transposed, 4KB
    __shared__ float    biasLds[32];     // 0 or -1e30 per key of chunk

    const int lane = threadIdx.x & 31;
    const int wave = threadIdx.x >> 5;
    const int lo = lane & 15;
    const int hi = lane >> 4;

    const int bh = blockIdx.y;
    const int b  = bh / HEADS;
    const int h  = bh - b * HEADS;
    const int q0 = blockIdx.x * 64 + wave * 16;

    const _Float16* qbase = qws + (size_t)bh * N * HEAD_DIM;
    const _Float16* kbase = kws + (size_t)bh * N * HEAD_DIM;
    const _Float16* vbase = vws + (size_t)bh * N * HEAD_DIM;

    // Q in B-layout (lane lo = query col, halves = d = e + 16*hi): contiguous 32B loads
    int qrow = q0 + lo; if (qrow >= N) qrow = N - 1;
    const v16h bq0 = *(const v16h*)(qbase + (size_t)qrow * HEAD_DIM + 16 * hi);
    const v16h bq1 = *(const v16h*)(qbase + (size_t)qrow * HEAD_DIM + 32 + 16 * hi);

    v8f o0 = {0,0,0,0,0,0,0,0}, o1 = o0, o2 = o0, o3 = o0;
    float m_run = -1e30f, l_run = 0.0f;

    const int nChunks = (N + 31) >> 5;
    for (int cidx = 0; cidx < nChunks; ++cidx) {
        const int k0 = cidx << 5;
        __syncthreads();
        // ---- cooperative staging: 128 threads x 16 halves for K and V ----
        {
            const int t = threadIdx.x;
            const int krel  = t >> 2;          // 0..31 key within chunk
            const int dpart = (t & 3) << 4;    // 0,16,32,48
            int key = k0 + krel; if (key >= N) key = N - 1;
            *(v16h*)(&Klds[krel * 64 + dpart]) =
                *(const v16h*)(kbase + (size_t)key * HEAD_DIM + dpart);
            const v16h vv = *(const v16h*)(vbase + (size_t)key * HEAD_DIM + dpart);
#pragma unroll
            for (int j = 0; j < 16; ++j) Vt[(dpart + j) * 32 + krel] = vv[j];
            if (t < 32) {
                const int kk = k0 + t;
                biasLds[t] = (kk < N && mask[(size_t)b * N + kk]) ? 0.0f : -1e30f;
            }
            // prefetch next chunk of K into cache
            if (k0 + 32 < N)
                __builtin_prefetch(kbase + (size_t)(k0 + 32 + krel) * HEAD_DIM + dpart, 0, 1);
        }
        __syncthreads();

        // ---- S^T = K x Q^T : 4 WMMAs ----
        const v16h a00 = loadA(&Klds[lo * 64],        hi, 0);
        const v16h a01 = loadA(&Klds[lo * 64],        hi, 1);
        const v16h a10 = loadA(&Klds[(lo + 16) * 64], hi, 0);
        const v16h a11 = loadA(&Klds[(lo + 16) * 64], hi, 1);

        v8f s0 = {0,0,0,0,0,0,0,0}, s1 = s0;
        s0 = wmma_f16(a00, bq0, s0);  s0 = wmma_f16(a01, bq1, s0);
        s1 = wmma_f16(a10, bq0, s1);  s1 = wmma_f16(a11, bq1, s1);

        // mask / OOB bias (key index of s0[i] is i+8*hi, of s1[i] is 16+i+8*hi)
#pragma unroll
        for (int i = 0; i < 8; ++i) {
            s0[i] += biasLds[(hi << 3) + i];
            s1[i] += biasLds[16 + (hi << 3) + i];
        }

        // ---- online softmax (per query = lane lo; halves exchanged via xor16) ----
        float mloc = -1e30f;
#pragma unroll
        for (int i = 0; i < 8; ++i) mloc = fmaxf(mloc, fmaxf(s0[i], s1[i]));
        mloc = fmaxf(mloc, bperm(mloc, lane ^ 16));
        const float m_new = fmaxf(m_run, mloc);
        const float alpha = __expf(m_run - m_new);

        v16h p;                      // lands directly in A-layout for PV
        float rs = 0.0f;
#pragma unroll
        for (int i = 0; i < 8; ++i) {
            const float e0 = __expf(s0[i] - m_new);
            const float e1 = __expf(s1[i] - m_new);
            rs += e0 + e1;
            p[i]     = (_Float16)e0;
            p[i + 8] = (_Float16)e1;
        }
        rs += bperm(rs, lane ^ 16);
        l_run = l_run * alpha + rs;
        m_run = m_new;

        // rescale O: row q of accumulator element i is i+8*hi -> broadcast alpha[q]
#pragma unroll
        for (int i = 0; i < 8; ++i) {
            const float av = bperm(alpha, (hi << 3) + i);
            o0[i] *= av; o1[i] *= av; o2[i] *= av; o3[i] *= av;
        }

        // ---- O += P x V : 4 WMMAs, B-operand reads from Vt as two contiguous 16B runs ----
        v16h bv0, bv1, bv2, bv3;
#pragma unroll
        for (int j = 0; j < 16; ++j) {
            const int kk = (j & 7) + 8 * hi + 16 * (j >> 3);
            bv0[j] = Vt[(0  + lo) * 32 + kk];
            bv1[j] = Vt[(16 + lo) * 32 + kk];
            bv2[j] = Vt[(32 + lo) * 32 + kk];
            bv3[j] = Vt[(48 + lo) * 32 + kk];
        }
        o0 = wmma_f16(p, bv0, o0);
        o1 = wmma_f16(p, bv1, o1);
        o2 = wmma_f16(p, bv2, o2);
        o3 = wmma_f16(p, bv3, o3);
    }

    // ---- epilogue: normalize by l and store (lane lo = d-col, vgpr i = q-row) ----
#pragma unroll
    for (int i = 0; i < 8; ++i) {
        const float lv  = bperm(l_run, (hi << 3) + i);
        const float inv = 1.0f / lv;
        const int qr = q0 + (hi << 3) + i;
        if (qr < N) {
            const size_t ob = ((size_t)b * N + qr) * HIDDEN + h * HEAD_DIM + lo;
            out[ob +  0] = o0[i] * inv;
            out[ob + 16] = o1[i] * inv;
            out[ob + 32] = o2[i] * inv;
            out[ob + 48] = o3[i] * inv;
        }
    }
}

// =====================================================================
extern "C" void kernel_launch(void* const* d_in, const int* in_sizes, int n_in,
                              void* d_out, int out_size, void* d_ws, size_t ws_size,
                              hipStream_t stream) {
    const float* q  = (const float*)d_in[0];
    const float* k  = (const float*)d_in[1];
    const float* v  = (const float*)d_in[2];
    const unsigned char* mask = (const unsigned char*)d_in[3];
    const float* qw = (const float*)d_in[4];
    const float* kw = (const float*)d_in[5];
    const int* rope_w     = (const int*)d_in[7];
    const int* rope_start = (const int*)d_in[8];
    const int* rope_len   = (const int*)d_in[9];

    const int B = BATCH;                 // fixed by the reference setup
    const int N = in_sizes[3] / B;       // mask has b*n elements

    size_t per = ((size_t)B * HEADS * N * HEAD_DIM * sizeof(_Float16) + 255) & ~(size_t)255;
    _Float16* qws = (_Float16*)d_ws;
    _Float16* kws = (_Float16*)((char*)d_ws + per);
    _Float16* vws = (_Float16*)((char*)d_ws + 2 * per);

    const int rows = B * N * HEADS;
    const int prepBlocks = (rows + 7) / 8;           // 8 waves / 256-thread block
    prep_kernel<<<prepBlocks, 256, 0, stream>>>(q, k, v, qw, kw,
                                                rope_start, rope_len, rope_w,
                                                qws, kws, vws, N);

    dim3 grid((N + 63) / 64, B * HEADS);             // 18 x 96 blocks
    attn_kernel<<<grid, 128, 0, stream>>>(qws, kws, vws, mask, (float*)d_out, N);
}
